// CrossAttention_6837587935785
// MI455X (gfx1250) — compile-verified, matrix operand order checked by reading, not run on hardware
//
#include <hip/hip_runtime.h>

// ---------------------------------------------------------------------------
// Types for CDNA5 WMMA (wave32): v_wmma_f32_16x16x32_bf16
// ---------------------------------------------------------------------------
typedef __bf16 bf16_t;
typedef __attribute__((ext_vector_type(4)))  __bf16 v4bf;
typedef __attribute__((ext_vector_type(8)))  __bf16 v8bf;
typedef __attribute__((ext_vector_type(16))) __bf16 v16bf;
typedef __attribute__((ext_vector_type(8)))  float  v8f;

#define WMMA_BF16(a, b, c) \
    __builtin_amdgcn_wmma_f32_16x16x32_bf16(false, (a), false, (b), (short)0, (c), false, false)

// Problem constants (B=2, S=2048, D=1024, H=16, dh=64)
#define SEQ    2048
#define DMODEL 1024
#define NHEAD  16
#define DHEAD  64
#define QK_SCALE 0.125f  // 64^-0.5

// Tensor Data Mover path (this toolchain: 6-arg builtin)
#if __has_builtin(__builtin_amdgcn_tensor_load_to_lds) && !defined(DISABLE_TDM)
#define USE_TDM 1
#else
#define USE_TDM 0
#endif

static __device__ __forceinline__ v8f vzero8f() {
    v8f z;
#pragma unroll
    for (int i = 0; i < 8; ++i) z[i] = 0.0f;
    return z;
}

// A/B fragment loader for 16x32 bf16 tiles with K-contiguous memory.
// ISA layout: lanes 0-15 hold M=lane, K = {0..7, 16..23}; lanes 16-31 hold
// M=lane-16, K = {8..15, 24..31}.  Two 16B loads per lane.
static __device__ __forceinline__ v16bf load_frag(const bf16_t* base, int pitch, int lane) {
    const int m  = lane & 15;
    const int kb = (lane >> 4) << 3;  // 0 or 8
    const bf16_t* p = base + m * pitch + kb;
    union { v16bf v; v8bf h[2]; } u;
    u.h[0] = *(const v8bf*)(p);
    u.h[1] = *(const v8bf*)(p + 16);
    return u.v;
}

#if USE_TDM
typedef unsigned int u32;
typedef __attribute__((ext_vector_type(4))) unsigned int v4u;
typedef __attribute__((ext_vector_type(8))) int         v8i;
typedef __attribute__((ext_vector_type(4))) int         v4i;

// Issue a TDM load of a 64x64 bf16 tile (row stride 64 elems) into LDS at
// lds_off, with DMA-side padding: 32 DWORDs (128B row) then +4 DWORDs pad
// -> effective LDS pitch 72 bf16 elements (matches load_frag pitch).
static __device__ __forceinline__ void tdm_load_tile64(const void* gptr, u32 lds_off) {
    const unsigned long long ga = (unsigned long long)gptr;
    v4u g0;
    g0[0] = 1u;                                       // count=1, user mode
    g0[1] = lds_off;                                  // LDS byte address
    g0[2] = (u32)(ga & 0xffffffffu);                  // global_addr[31:0]
    g0[3] = (u32)((ga >> 32) & 0x01ffffffu) | (2u << 30);  // addr[56:32] | type=2
    v8i g1;
    g1[0] = (int)((1u << 16)        // data_size = 2 bytes
                | (1u << 20)        // pad_enable
                | (4u << 22)        // pad_interval: 32 DWORDs
                | (3u << 25));      // pad_amount:   4 DWORDs
    g1[1] = (int)(64u << 16);                         // tensor_dim0 lo16 = 64
    g1[2] = (int)(64u << 16);                         // dim0 hi16=0 | tensor_dim1 lo16 = 64
    g1[3] = (int)(64u << 16);                         // dim1 hi16=0 | tile_dim0 = 64
    g1[4] = (int)(64u);                               // tile_dim1 = 64 | tile_dim2 = 0
    g1[5] = (int)(64u);                               // tensor_dim0_stride = 64 elems
    g1[6] = 0;
    g1[7] = 0;
    v4i z4; z4[0] = 0; z4[1] = 0; z4[2] = 0; z4[3] = 0;
    v8i z8;
#pragma unroll
    for (int i = 0; i < 8; ++i) z8[i] = 0;
    __builtin_amdgcn_tensor_load_to_lds(g0, g1, z4, z4, z8, 0);
}
#endif

// ---------------------------------------------------------------------------
// Kernel 1: fused QKV projection.  C = X @ W^T  (X:[4096,1024] f32, W:[N,K] f32)
// grid = (N/128, M/128, 3);  z selects {Wq,Wk,Wv}.  Register double-buffered
// global->LDS staging; output scattered bf16 into [B,H,S,dh]; SCALE folded
// into Q.
// ---------------------------------------------------------------------------
__global__ __launch_bounds__(256) void qkv_gemm(
    const float* __restrict__ X,
    const float* __restrict__ Wq, const float* __restrict__ Wk, const float* __restrict__ Wv,
    bf16_t* __restrict__ Qo, bf16_t* __restrict__ Ko, bf16_t* __restrict__ Vo)
{
    const int Kd = DMODEL;
    const float* W = (blockIdx.z == 0) ? Wq : (blockIdx.z == 1 ? Wk : Wv);
    bf16_t*      O = (blockIdx.z == 0) ? Qo : (blockIdx.z == 1 ? Ko : Vo);
    const float scale = (blockIdx.z == 0) ? QK_SCALE : 1.0f;

    const int m0 = blockIdx.y * 128;
    const int n0 = blockIdx.x * 128;
    const int tid  = threadIdx.x;
    const int lane = tid & 31;
    const int wave = tid >> 5;
    const int wm = wave & 1;   // 64 rows each
    const int wn = wave >> 1;  // 32 cols each

    __shared__ bf16_t Als[128 * 40];  // pitch 40 bf16 (80B, 16B aligned)
    __shared__ bf16_t Bls[128 * 40];

    v8f acc[4][2];
#pragma unroll
    for (int i = 0; i < 4; ++i) { acc[i][0] = vzero8f(); acc[i][1] = vzero8f(); }

    float4 fa[4], fb[4];
    auto load_regs = [&](int k0) {
#pragma unroll
        for (int it = 0; it < 4; ++it) {
            const int idx = tid + it * 256;
            const int r  = idx >> 3;
            const int cg = (idx & 7) << 2;
            fa[it] = *(const float4*)(X + (size_t)(m0 + r) * Kd + k0 + cg);
            fb[it] = *(const float4*)(W + (size_t)(n0 + r) * Kd + k0 + cg);
        }
    };
    auto store_lds = [&]() {
#pragma unroll
        for (int it = 0; it < 4; ++it) {
            const int idx = tid + it * 256;
            const int r  = idx >> 3;
            const int cg = (idx & 7) << 2;
            v4bf pa = { (bf16_t)fa[it].x, (bf16_t)fa[it].y, (bf16_t)fa[it].z, (bf16_t)fa[it].w };
            *(v4bf*)&Als[r * 40 + cg] = pa;
            v4bf pb = { (bf16_t)fb[it].x, (bf16_t)fb[it].y, (bf16_t)fb[it].z, (bf16_t)fb[it].w };
            *(v4bf*)&Bls[r * 40 + cg] = pb;
        }
    };

    load_regs(0);
    for (int kt = 0; kt < Kd / 32; ++kt) {
        __syncthreads();
        store_lds();
        __syncthreads();
        if (kt + 1 < Kd / 32) load_regs((kt + 1) * 32);  // overlap next tile with WMMA

        // Load all fragments first, then issue WMMAs back-to-back.
        const v16bf b0 = load_frag(&Bls[(wn * 32 + 0)  * 40], 40, lane);
        const v16bf b1 = load_frag(&Bls[(wn * 32 + 16) * 40], 40, lane);
        const v16bf a0 = load_frag(&Als[(wm * 64 +  0) * 40], 40, lane);
        const v16bf a1 = load_frag(&Als[(wm * 64 + 16) * 40], 40, lane);
        const v16bf a2 = load_frag(&Als[(wm * 64 + 32) * 40], 40, lane);
        const v16bf a3 = load_frag(&Als[(wm * 64 + 48) * 40], 40, lane);
        acc[0][0] = WMMA_BF16(a0, b0, acc[0][0]);
        acc[1][0] = WMMA_BF16(a1, b0, acc[1][0]);
        acc[2][0] = WMMA_BF16(a2, b0, acc[2][0]);
        acc[3][0] = WMMA_BF16(a3, b0, acc[3][0]);
        acc[0][1] = WMMA_BF16(a0, b1, acc[0][1]);
        acc[1][1] = WMMA_BF16(a1, b1, acc[1][1]);
        acc[2][1] = WMMA_BF16(a2, b1, acc[2][1]);
        acc[3][1] = WMMA_BF16(a3, b1, acc[3][1]);
    }

    // Scatter to [B,H,S,dh] bf16.
#pragma unroll
    for (int mi = 0; mi < 4; ++mi)
#pragma unroll
        for (int ni = 0; ni < 2; ++ni)
#pragma unroll
            for (int r = 0; r < 8; ++r) {
                const int m = m0 + wm * 64 + mi * 16 + r + ((lane >> 4) << 3);
                const int n = n0 + wn * 32 + ni * 16 + (lane & 15);
                const int b = m >> 11, s = m & (SEQ - 1);
                const int h = n >> 6, d = n & (DHEAD - 1);
                O[(((size_t)b * NHEAD + h) * SEQ + s) * DHEAD + d] =
                    (bf16_t)(acc[mi][ni][r] * scale);
            }
}

// ---------------------------------------------------------------------------
// Kernel 2: flash attention.  grid = (S/128, B*H), 256 threads (8 waves),
// each wave owns 16 query rows; loop over 64-key blocks with online softmax.
// K block staged by the Tensor Data Mover (DMA pads rows to pitch 72); V block
// staged transposed by the VALU path.
// ---------------------------------------------------------------------------
__global__ __launch_bounds__(256) void attn_kernel(
    const bf16_t* __restrict__ Q, const bf16_t* __restrict__ K,
    const bf16_t* __restrict__ V, bf16_t* __restrict__ O)
{
    const int bh   = blockIdx.y;           // b*16 + h
    const int tid  = threadIdx.x;
    const int lane = tid & 31;
    const int wave = tid >> 5;
    const int qrow0 = blockIdx.x * 128 + wave * 16;

    const bf16_t* Qh = Q + (size_t)bh * SEQ * DHEAD;
    const bf16_t* Kh = K + (size_t)bh * SEQ * DHEAD;
    const bf16_t* Vh = V + (size_t)bh * SEQ * DHEAD;

    __shared__ bf16_t Kls[64 * 72];        // key-major, dh contiguous (pitch 72)
    __shared__ bf16_t Vt [64 * 72];        // dh-major (transposed), key contiguous
    __shared__ bf16_t Pls[8 * 16 * 72];    // per-wave P staging

    // Q fragments (16 rows x dh=64) straight from global (K-contiguous 16B loads)
    const v16bf qa0 = load_frag(Qh + (size_t)qrow0 * DHEAD + 0,  DHEAD, lane);
    const v16bf qa1 = load_frag(Qh + (size_t)qrow0 * DHEAD + 32, DHEAD, lane);

    v8f oacc[4];
#pragma unroll
    for (int t = 0; t < 4; ++t) oacc[t] = vzero8f();
    float mrow[8], lrow[8];
#pragma unroll
    for (int r = 0; r < 8; ++r) { mrow[r] = -1e30f; lrow[r] = 0.0f; }

    for (int j = 0; j < SEQ / 64; ++j) {
        __syncthreads();   // previous block fully consumed

#if USE_TDM
        // DMA the 64x64 K tile into LDS (pads each 128B row by 16B -> pitch 72)
        if (wave == 0) {
            tdm_load_tile64(Kh + (size_t)(j * 64) * DHEAD,
                            (u32)(unsigned long long)(const void*)&Kls[0]);
        }
#endif
        // Stage V transposed (and K manually when no TDM).
#pragma unroll
        for (int it = 0; it < 2; ++it) {
            const int idx = tid + it * 256;   // 0..511
            const int r  = idx >> 3;          // key row 0..63
            const int cg = (idx & 7) << 3;    // dh 0..56
#if !USE_TDM
            const v8bf kv = *(const v8bf*)(Kh + (size_t)(j * 64 + r) * DHEAD + cg);
            *(v8bf*)&Kls[r * 72 + cg] = kv;
#endif
            const v8bf vv = *(const v8bf*)(Vh + (size_t)(j * 64 + r) * DHEAD + cg);
#pragma unroll
            for (int e = 0; e < 8; ++e) Vt[(cg + e) * 72 + r] = vv[e];
        }
#if USE_TDM
        if (wave == 0) __builtin_amdgcn_s_wait_tensorcnt(0);
#endif
        __syncthreads();

        // Warm caches for the next key block while we compute on this one.
        if (j + 1 < SEQ / 64) {
            __builtin_prefetch(Kh + (size_t)((j + 1) * 64 + (tid >> 2)) * DHEAD + ((tid & 3) << 4), 0, 1);
            __builtin_prefetch(Vh + (size_t)((j + 1) * 64 + (tid >> 2)) * DHEAD + ((tid & 3) << 4), 0, 1);
        }

        // scores S = Q . K^T : 4 column tiles of 16 keys, K-dim = dh = 64
        v8f sc[4];
#pragma unroll
        for (int n = 0; n < 4; ++n) {
            const v16bf kb0 = load_frag(&Kls[n * 16 * 72 + 0],  72, lane);
            const v16bf kb1 = load_frag(&Kls[n * 16 * 72 + 32], 72, lane);
            v8f s = vzero8f();
            s = WMMA_BF16(qa0, kb0, s);
            s = WMMA_BF16(qa1, kb1, s);
            sc[n] = s;
        }

        // Online softmax: per-row max over 64 keys (lanes within 16-halves)
        float al[8];
#pragma unroll
        for (int r = 0; r < 8; ++r) {
            float mx = fmaxf(fmaxf(sc[0][r], sc[1][r]), fmaxf(sc[2][r], sc[3][r]));
#pragma unroll
            for (int off = 1; off < 16; off <<= 1) mx = fmaxf(mx, __shfl_xor(mx, off, 32));
            const float mnew = fmaxf(mrow[r], mx);
            al[r]  = __expf(mrow[r] - mnew);
            mrow[r] = mnew;
            lrow[r] *= al[r];
        }
#pragma unroll
        for (int t = 0; t < 4; ++t)
#pragma unroll
            for (int r = 0; r < 8; ++r) oacc[t][r] *= al[r];

        // P = exp(S - m); write bf16 to per-wave LDS for A-layout reload
        bf16_t* Pw = &Pls[wave * 16 * 72];
        float rsum[8];
#pragma unroll
        for (int r = 0; r < 8; ++r) rsum[r] = 0.0f;
#pragma unroll
        for (int n = 0; n < 4; ++n)
#pragma unroll
            for (int r = 0; r < 8; ++r) {
                const float p = __expf(sc[n][r] - mrow[r]);
                rsum[r] += p;
                const int row = r + ((lane >> 4) << 3);
                Pw[row * 72 + n * 16 + (lane & 15)] = (bf16_t)p;
            }
#pragma unroll
        for (int r = 0; r < 8; ++r) {
            float s = rsum[r];
#pragma unroll
            for (int off = 1; off < 16; off <<= 1) s += __shfl_xor(s, off, 32);
            lrow[r] += s;
        }

        // Wave-local LDS RAW fence (CDNA5 split counter)
        asm volatile("s_wait_dscnt 0" ::: "memory");

        // O += P . V : load all fragments, then WMMAs back-to-back
        const v16bf pa0 = load_frag(Pw + 0,  72, lane);
        const v16bf pa1 = load_frag(Pw + 32, 72, lane);
        const v16bf vb00 = load_frag(&Vt[ 0 * 72 + 0],  72, lane);
        const v16bf vb01 = load_frag(&Vt[ 0 * 72 + 32], 72, lane);
        const v16bf vb10 = load_frag(&Vt[16 * 72 + 0],  72, lane);
        const v16bf vb11 = load_frag(&Vt[16 * 72 + 32], 72, lane);
        const v16bf vb20 = load_frag(&Vt[32 * 72 + 0],  72, lane);
        const v16bf vb21 = load_frag(&Vt[32 * 72 + 32], 72, lane);
        const v16bf vb30 = load_frag(&Vt[48 * 72 + 0],  72, lane);
        const v16bf vb31 = load_frag(&Vt[48 * 72 + 32], 72, lane);
        oacc[0] = WMMA_BF16(pa0, vb00, oacc[0]);
        oacc[1] = WMMA_BF16(pa0, vb10, oacc[1]);
        oacc[2] = WMMA_BF16(pa0, vb20, oacc[2]);
        oacc[3] = WMMA_BF16(pa0, vb30, oacc[3]);
        oacc[0] = WMMA_BF16(pa1, vb01, oacc[0]);
        oacc[1] = WMMA_BF16(pa1, vb11, oacc[1]);
        oacc[2] = WMMA_BF16(pa1, vb21, oacc[2]);
        oacc[3] = WMMA_BF16(pa1, vb31, oacc[3]);
    }

    // Normalize and write O in [B,S,D] bf16 (head-recombined)
    const int b = bh >> 4, h = bh & 15;
#pragma unroll
    for (int t = 0; t < 4; ++t)
#pragma unroll
        for (int r = 0; r < 8; ++r) {
            const int row = qrow0 + r + ((lane >> 4) << 3);
            const int col = h * DHEAD + t * 16 + (lane & 15);
            O[((size_t)b * SEQ + row) * DMODEL + col] = (bf16_t)(oacc[t][r] / lrow[r]);
        }
}

// ---------------------------------------------------------------------------
// Kernel 3: output projection.  out = O @ Wo^T + bo  (O bf16, out fp32)
// ---------------------------------------------------------------------------
__global__ __launch_bounds__(256) void out_gemm(
    const bf16_t* __restrict__ A, const float* __restrict__ W,
    const float* __restrict__ bias, float* __restrict__ Cout)
{
    const int Kd = DMODEL;
    const int m0 = blockIdx.y * 128;
    const int n0 = blockIdx.x * 128;
    const int tid  = threadIdx.x;
    const int lane = tid & 31;
    const int wave = tid >> 5;
    const int wm = wave & 1;
    const int wn = wave >> 1;

    __shared__ bf16_t Als[128 * 40];
    __shared__ bf16_t Bls[128 * 40];

    v8f acc[4][2];
#pragma unroll
    for (int i = 0; i < 4; ++i) { acc[i][0] = vzero8f(); acc[i][1] = vzero8f(); }

    v8bf   av[2];
    float4 fb[4];
    auto load_regs = [&](int k0) {
#pragma unroll
        for (int it = 0; it < 2; ++it) {
            const int idx = tid + it * 256;
            const int r  = idx >> 2;
            const int cg = (idx & 3) << 3;
            av[it] = *(const v8bf*)(A + (size_t)(m0 + r) * Kd + k0 + cg);
        }
#pragma unroll
        for (int it = 0; it < 4; ++it) {
            const int idx = tid + it * 256;
            const int r  = idx >> 3;
            const int cg = (idx & 7) << 2;
            fb[it] = *(const float4*)(W + (size_t)(n0 + r) * Kd + k0 + cg);
        }
    };
    auto store_lds = [&]() {
#pragma unroll
        for (int it = 0; it < 2; ++it) {
            const int idx = tid + it * 256;
            const int r  = idx >> 2;
            const int cg = (idx & 3) << 3;
            *(v8bf*)&Als[r * 40 + cg] = av[it];
        }
#pragma unroll
        for (int it = 0; it < 4; ++it) {
            const int idx = tid + it * 256;
            const int r  = idx >> 3;
            const int cg = (idx & 7) << 2;
            v4bf pb = { (bf16_t)fb[it].x, (bf16_t)fb[it].y, (bf16_t)fb[it].z, (bf16_t)fb[it].w };
            *(v4bf*)&Bls[r * 40 + cg] = pb;
        }
    };

    load_regs(0);
    for (int kt = 0; kt < Kd / 32; ++kt) {
        __syncthreads();
        store_lds();
        __syncthreads();
        if (kt + 1 < Kd / 32) load_regs((kt + 1) * 32);

        const v16bf b0 = load_frag(&Bls[(wn * 32 + 0)  * 40], 40, lane);
        const v16bf b1 = load_frag(&Bls[(wn * 32 + 16) * 40], 40, lane);
        const v16bf a0 = load_frag(&Als[(wm * 64 +  0) * 40], 40, lane);
        const v16bf a1 = load_frag(&Als[(wm * 64 + 16) * 40], 40, lane);
        const v16bf a2 = load_frag(&Als[(wm * 64 + 32) * 40], 40, lane);
        const v16bf a3 = load_frag(&Als[(wm * 64 + 48) * 40], 40, lane);
        acc[0][0] = WMMA_BF16(a0, b0, acc[0][0]);
        acc[1][0] = WMMA_BF16(a1, b0, acc[1][0]);
        acc[2][0] = WMMA_BF16(a2, b0, acc[2][0]);
        acc[3][0] = WMMA_BF16(a3, b0, acc[3][0]);
        acc[0][1] = WMMA_BF16(a0, b1, acc[0][1]);
        acc[1][1] = WMMA_BF16(a1, b1, acc[1][1]);
        acc[2][1] = WMMA_BF16(a2, b1, acc[2][1]);
        acc[3][1] = WMMA_BF16(a3, b1, acc[3][1]);
    }

#pragma unroll
    for (int mi = 0; mi < 4; ++mi)
#pragma unroll
        for (int ni = 0; ni < 2; ++ni)
#pragma unroll
            for (int r = 0; r < 8; ++r) {
                const int m = m0 + wm * 64 + mi * 16 + r + ((lane >> 4) << 3);
                const int n = n0 + wn * 32 + ni * 16 + (lane & 15);
                Cout[(size_t)m * DMODEL + n] = acc[mi][ni][r] + bias[n];
            }
}

// ---------------------------------------------------------------------------
// Host launcher
// ---------------------------------------------------------------------------
extern "C" void kernel_launch(void* const* d_in, const int* in_sizes, int n_in,
                              void* d_out, int out_size, void* d_ws, size_t ws_size,
                              hipStream_t stream) {
    const float* X  = (const float*)d_in[0];   // hidden_states [2,2048,1024]
    const float* Wq = (const float*)d_in[1];
    const float* Wk = (const float*)d_in[2];
    const float* Wv = (const float*)d_in[3];
    const float* Wo = (const float*)d_in[4];
    const float* bo = (const float*)d_in[5];
    float* out = (float*)d_out;

    const size_t qelems = (size_t)2 * SEQ * DMODEL;   // per-tensor bf16 elements
    bf16_t* Qb = (bf16_t*)d_ws;
    bf16_t* Kb = Qb + qelems;
    bf16_t* Vb = Kb + qelems;
    bf16_t* Ob = Vb + qelems;

    // 1) Q/K/V projections (z = 0/1/2), bf16, [B,H,S,dh]; SCALE folded into Q.
    qkv_gemm<<<dim3(DMODEL / 128, (2 * SEQ) / 128, 3), 256, 0, stream>>>(
        X, Wq, Wk, Wv, Qb, Kb, Vb);

    // 2) Flash attention -> O bf16 in [B,S,D]
    attn_kernel<<<dim3(SEQ / 128, 2 * NHEAD), 256, 0, stream>>>(Qb, Kb, Vb, Ob);

    // 3) Output projection + bias -> fp32 d_out
    out_gemm<<<dim3(DMODEL / 128, (2 * SEQ) / 128), 256, 0, stream>>>(Ob, Wo, bo, out);
}
